// Seq2seqAttn_12086037971160
// MI455X (gfx1250) — matrix operand order repaired and checked by reading, not compiled, not run on hardware
//
#include <hip/hip_runtime.h>

// ---------------- problem constants ----------------
#define Bc  256   // batch
#define Sc  512   // encoder sequence length
#define Tc  32    // decode horizon
#define DIc 5     // feature dim
#define Hc  256   // hidden
#define Gc  1024  // 4*H gate dim

typedef _Float16 h8   __attribute__((ext_vector_type(8)));
typedef _Float16 v16h __attribute__((ext_vector_type(16)));
typedef float    v8f  __attribute__((ext_vector_type(8)));

// ---------------- WMMA helpers (V_WMMA_F32_16X16X32_F16) ----------------
// C/D layout: vgpr r, lane L  ->  (m = r + (L>=16)*8, n = L%16)
// A layout  : lane L -> row m=L%16; halfs 0..7 = K kb..kb+7, 8..15 = K kb+16..kb+23, kb = (L>=16)*8
// B layout  : lane L -> col n=L%16; halfs 0..7 = K kb..kb+7, 8..15 = K kb+16..kb+23, kb = (L>=16)*8; B[k][n] = W[n][k]
__device__ __forceinline__ v8f wmma_f32(v16h a, v16h b, v8f c) {
  return __builtin_amdgcn_wmma_f32_16x16x32_f16(false, a, false, b, (short)0, c, false, false);
}

__device__ __forceinline__ v16h mk16(h8 lo, h8 hi) {
  v16h r;
#pragma unroll
  for (int i = 0; i < 8; ++i) { r[i] = lo[i]; r[i + 8] = hi[i]; }
  return r;
}

// A operand from LDS f16 tile [16][256] row-major; 16B-aligned b128 loads
__device__ __forceinline__ v16h load_A(const _Float16* hA, int lm, int kt, int laneHi) {
  const _Float16* p = hA + lm * Hc + kt * 32 + laneHi * 8;
  h8 lo = *(const h8*)p;
  h8 hi = *(const h8*)(p + 16);
  return mk16(lo, hi);
}

// B operand from pre-swizzled global weights: tile (nt,kt), 32B contiguous per lane
__device__ __forceinline__ v16h load_B(const _Float16* swz, int nt, int kt, int lane) {
  const _Float16* p = swz + (size_t)(((nt * 8 + kt) * 32) + lane) * 16;
  h8 lo = *(const h8*)p;
  h8 hi = *(const h8*)(p + 8);
  return mk16(lo, hi);
}

__device__ __forceinline__ float sigm(float x) { return 1.f / (1.f + __expf(-x)); }

// ---------------- prep: swizzle W[1024][256] f32 -> WMMA-B f16 layout ----------------
__global__ void __launch_bounds__(256) swz_kernel(const float* __restrict__ src,
                                                  _Float16* __restrict__ dst) {
  int gid = blockIdx.x * 256 + threadIdx.x;          // one 16-half group per thread
  if (gid >= (Gc / 16) * (Hc / 32) * 32) return;     // 16384 groups
  int lane = gid & 31;
  int kt   = (gid >> 5) & 7;
  int nt   = gid >> 8;
  int n  = nt * 16 + (lane & 15);
  int kb = kt * 32 + ((lane >> 4) ? 8 : 0);
#pragma unroll
  for (int i = 0; i < 16; ++i)
    dst[(size_t)gid * 16 + i] = (_Float16)src[(size_t)n * Hc + kb + (i & 7) + (i >> 3) * 16];
}

__global__ void init_pred_kernel(float* pred) {
  int i = blockIdx.x * blockDim.x + threadIdx.x;
  if (i < Bc * DIc) pred[i] = 1.f;
}

// ---------------- encoder: 2 stacked LSTMs, 512 steps, 16 batch rows / WG ----------------
__global__ void __launch_bounds__(256)
enc_kernel(const float* __restrict__ srcs,
           const float* __restrict__ wih_e1, const float* __restrict__ b_e1,
           const float* __restrict__ b_e2,
           const _Float16* __restrict__ whh1s, const _Float16* __restrict__ wih2s,
           const _Float16* __restrict__ whh2s,
           _Float16* __restrict__ enc_out,
           float* __restrict__ h_state, float* __restrict__ c_state) {
  __shared__ __align__(16) _Float16 hA1[16 * Hc];   // layer1 h (f16, A operand)
  __shared__ __align__(16) _Float16 hA2[16 * Hc];   // layer2 h
  __shared__ float srcT[16 * DIc];
  __shared__ float w1[Gc * DIc];                    // wih_e1 cache (K=5, VALU path)
  __shared__ float bb1[Gc];
  __shared__ float bb2[Gc];

  const int tid = threadIdx.x;
  const int lane = tid & 31, wave = tid >> 5;
  const int lm = lane & 15, laneHi = lane >> 4;
  const int b0 = blockIdx.x * 16;

  for (int i = tid; i < Gc * DIc; i += 256) w1[i] = wih_e1[i];
  for (int i = tid; i < Gc; i += 256) { bb1[i] = b_e1[i]; bb2[i] = b_e2[i]; }
  for (int i = tid; i < 16 * Hc; i += 256) { hA1[i] = (_Float16)0.f; hA2[i] = (_Float16)0.f; }

  float c1[2][8], c2[2][8];
#pragma unroll
  for (int jt = 0; jt < 2; ++jt)
#pragma unroll
    for (int r = 0; r < 8; ++r) { c1[jt][r] = 0.f; c2[jt][r] = 0.f; }
  __syncthreads();

  for (int t = 0; t < Sc; ++t) {
    if (tid < 16 * DIc) {
      int r = tid / DIc, d = tid % DIc;
      srcT[tid] = srcs[((size_t)(b0 + r) * Sc + t) * DIc + d];
    }
    __syncthreads();

    // ---- layer 1: acc = b1 + x@Wih1^T (K=5, VALU), then += h1@Whh1^T (WMMA K=256)
    v8f acc[4][2];
#pragma unroll
    for (int g = 0; g < 4; ++g)
#pragma unroll
      for (int jt = 0; jt < 2; ++jt) {
        int n = (g * 16 + 2 * wave + jt) * 16 + lm;
        float q0 = w1[n * DIc], q1 = w1[n * DIc + 1], q2 = w1[n * DIc + 2],
              q3 = w1[n * DIc + 3], q4 = w1[n * DIc + 4];
        float bn = bb1[n];
#pragma unroll
        for (int r = 0; r < 8; ++r) {
          int m = r + laneHi * 8;
          acc[g][jt][r] = bn + srcT[m * DIc] * q0 + srcT[m * DIc + 1] * q1 +
                          srcT[m * DIc + 2] * q2 + srcT[m * DIc + 3] * q3 +
                          srcT[m * DIc + 4] * q4;
        }
      }
    for (int kt = 0; kt < 8; ++kt) {
      v16h a = load_A(hA1, lm, kt, laneHi);
#pragma unroll
      for (int g = 0; g < 4; ++g)
#pragma unroll
        for (int jt = 0; jt < 2; ++jt)
          acc[g][jt] = wmma_f32(a, load_B(whh1s, g * 16 + 2 * wave + jt, kt, lane), acc[g][jt]);
    }
    __syncthreads();  // all hA1 reads complete before rewrite
    // pointwise LSTM update: all four gates for (m,j) live in this lane's regs
#pragma unroll
    for (int jt = 0; jt < 2; ++jt)
#pragma unroll
      for (int r = 0; r < 8; ++r) {
        int m = r + laneHi * 8;
        int j = (2 * wave + jt) * 16 + lm;
        float iv = sigm(acc[0][jt][r]), fv = sigm(acc[1][jt][r]);
        float gv = tanhf(acc[2][jt][r]), ov = sigm(acc[3][jt][r]);
        float cn = fv * c1[jt][r] + iv * gv;
        c1[jt][r] = cn;
        hA1[m * Hc + j] = (_Float16)(ov * tanhf(cn));
      }
    __syncthreads();

    // ---- layer 2: acc = b2 + h1@Wih2^T + h2@Whh2^T (both WMMA K=256)
#pragma unroll
    for (int g = 0; g < 4; ++g)
#pragma unroll
      for (int jt = 0; jt < 2; ++jt) {
        float bn = bb2[(g * 16 + 2 * wave + jt) * 16 + lm];
#pragma unroll
        for (int r = 0; r < 8; ++r) acc[g][jt][r] = bn;
      }
    for (int kt = 0; kt < 8; ++kt) {
      v16h a = load_A(hA1, lm, kt, laneHi);
#pragma unroll
      for (int g = 0; g < 4; ++g)
#pragma unroll
        for (int jt = 0; jt < 2; ++jt)
          acc[g][jt] = wmma_f32(a, load_B(wih2s, g * 16 + 2 * wave + jt, kt, lane), acc[g][jt]);
    }
    for (int kt = 0; kt < 8; ++kt) {
      v16h a = load_A(hA2, lm, kt, laneHi);
#pragma unroll
      for (int g = 0; g < 4; ++g)
#pragma unroll
        for (int jt = 0; jt < 2; ++jt)
          acc[g][jt] = wmma_f32(a, load_B(whh2s, g * 16 + 2 * wave + jt, kt, lane), acc[g][jt]);
    }
    __syncthreads();  // all hA2 reads complete before rewrite
#pragma unroll
    for (int jt = 0; jt < 2; ++jt)
#pragma unroll
      for (int r = 0; r < 8; ++r) {
        int m = r + laneHi * 8;
        int j = (2 * wave + jt) * 16 + lm;
        float iv = sigm(acc[0][jt][r]), fv = sigm(acc[1][jt][r]);
        float gv = tanhf(acc[2][jt][r]), ov = sigm(acc[3][jt][r]);
        float cn = fv * c2[jt][r] + iv * gv;
        c2[jt][r] = cn;
        float hv = ov * tanhf(cn);
        hA2[m * Hc + j] = (_Float16)hv;
        enc_out[((size_t)(b0 + m) * Sc + t) * Hc + j] = (_Float16)hv;  // L2-resident (64MB)
        if (t == Sc - 1) {
          h_state[(size_t)(b0 + m) * Hc + j] = hv;
          c_state[(size_t)(b0 + m) * Hc + j] = cn;
        }
      }
    __syncthreads();
  }
}

// ---------------- decoder step part 1: two cells + attention energies ----------------
__global__ void __launch_bounds__(256)
dec_cell_kernel(const float* __restrict__ wdi, const float* __restrict__ bdi,
                const float* __restrict__ b_d1, const float* __restrict__ b_d2,
                const _Float16* __restrict__ wd1s, const _Float16* __restrict__ whd1s,
                const _Float16* __restrict__ wd2s,
                const _Float16* __restrict__ enc_out,
                float* __restrict__ h_state, float* __restrict__ c_state,
                float* __restrict__ h2dec, float* __restrict__ energ,
                const float* __restrict__ pred) {
  __shared__ __align__(16) _Float16 xA[16 * Hc];    // projected decoder input
  __shared__ __align__(16) _Float16 hPA[16 * Hc];   // carried h (= prev h2)
  __shared__ __align__(16) _Float16 h1A[16 * Hc];   // cell1 output
  __shared__ float h2L[16 * Hc];                    // cell2 output (f32 for energies)
  __shared__ float predT[16 * DIc];

  const int tid = threadIdx.x;
  const int lane = tid & 31, wave = tid >> 5;
  const int lm = lane & 15, laneHi = lane >> 4;
  const int b0 = blockIdx.x * 16;

  if (tid < 16 * DIc) predT[tid] = pred[(size_t)(b0 + tid / DIc) * DIc + tid % DIc];
  __syncthreads();
  {  // x = pred@wdi^T + bdi (K=5 VALU) ; stage carried h as f16
    int j = tid;
    float q0 = wdi[j * DIc], q1 = wdi[j * DIc + 1], q2 = wdi[j * DIc + 2],
          q3 = wdi[j * DIc + 3], q4 = wdi[j * DIc + 4];
    float bj = bdi[j];
    for (int r = 0; r < 16; ++r) {
      float xv = bj + predT[r * DIc] * q0 + predT[r * DIc + 1] * q1 +
                 predT[r * DIc + 2] * q2 + predT[r * DIc + 3] * q3 + predT[r * DIc + 4] * q4;
      xA[r * Hc + j]  = (_Float16)xv;
      hPA[r * Hc + j] = (_Float16)h_state[(size_t)(b0 + r) * Hc + j];
    }
  }
  __syncthreads();

  // cell1: acc = b_d1 + x@Wih_d1^T + h@Whh_d1^T
  v8f acc[4][2];
#pragma unroll
  for (int g = 0; g < 4; ++g)
#pragma unroll
    for (int jt = 0; jt < 2; ++jt) {
      float bn = b_d1[(g * 16 + 2 * wave + jt) * 16 + lm];
#pragma unroll
      for (int r = 0; r < 8; ++r) acc[g][jt][r] = bn;
    }
  for (int kt = 0; kt < 8; ++kt) {
    v16h a = load_A(xA, lm, kt, laneHi);
#pragma unroll
    for (int g = 0; g < 4; ++g)
#pragma unroll
      for (int jt = 0; jt < 2; ++jt)
        acc[g][jt] = wmma_f32(a, load_B(wd1s, g * 16 + 2 * wave + jt, kt, lane), acc[g][jt]);
  }
  for (int kt = 0; kt < 8; ++kt) {
    v16h a = load_A(hPA, lm, kt, laneHi);
#pragma unroll
    for (int g = 0; g < 4; ++g)
#pragma unroll
      for (int jt = 0; jt < 2; ++jt)
        acc[g][jt] = wmma_f32(a, load_B(whd1s, g * 16 + 2 * wave + jt, kt, lane), acc[g][jt]);
  }
  // cell1 update (c1 only feeds h1; writes go to distinct buffer h1A)
#pragma unroll
  for (int jt = 0; jt < 2; ++jt)
#pragma unroll
    for (int r = 0; r < 8; ++r) {
      int m = r + laneHi * 8;
      int j = (2 * wave + jt) * 16 + lm;
      float iv = sigm(acc[0][jt][r]), fv = sigm(acc[1][jt][r]);
      float gv = tanhf(acc[2][jt][r]), ov = sigm(acc[3][jt][r]);
      float cp = c_state[(size_t)(b0 + m) * Hc + j];
      float cn = fv * cp + iv * gv;
      h1A[m * Hc + j] = (_Float16)(ov * tanhf(cn));
    }
  __syncthreads();

  // cell2 (zero initial state): acc = b_d2 + h1@Wih_d2^T
#pragma unroll
  for (int g = 0; g < 4; ++g)
#pragma unroll
    for (int jt = 0; jt < 2; ++jt) {
      float bn = b_d2[(g * 16 + 2 * wave + jt) * 16 + lm];
#pragma unroll
      for (int r = 0; r < 8; ++r) acc[g][jt][r] = bn;
    }
  for (int kt = 0; kt < 8; ++kt) {
    v16h a = load_A(h1A, lm, kt, laneHi);
#pragma unroll
    for (int g = 0; g < 4; ++g)
#pragma unroll
      for (int jt = 0; jt < 2; ++jt)
        acc[g][jt] = wmma_f32(a, load_B(wd2s, g * 16 + 2 * wave + jt, kt, lane), acc[g][jt]);
  }
#pragma unroll
  for (int jt = 0; jt < 2; ++jt)
#pragma unroll
    for (int r = 0; r < 8; ++r) {
      int m = r + laneHi * 8;
      int j = (2 * wave + jt) * 16 + lm;
      float iv = sigm(acc[0][jt][r]), fv = sigm(acc[1][jt][r]);
      float gv = tanhf(acc[2][jt][r]), ov = sigm(acc[3][jt][r]);
      (void)fv;  // c_prev is zero for cell2
      float cn = iv * gv;
      float hv = ov * tanhf(cn);
      h2L[m * Hc + j] = hv;
      h_state[(size_t)(b0 + m) * Hc + j] = hv;   // carry h2
      c_state[(size_t)(b0 + m) * Hc + j] = cn;   // carry c2
      h2dec[(size_t)(b0 + m) * Hc + j] = hv;     // for output linears
    }
  __syncthreads();

  // energies[b][s] = h2[b]·enc[b][s]  -> stored transposed [S][B] for batch softmax
  int r = tid >> 4, sb = tid & 15;
  const float* hrow = &h2L[r * Hc];
  for (int k = 0; k < 32; ++k) {
    int s = sb + k * 16;
    const _Float16* ep = enc_out + ((size_t)(b0 + r) * Sc + s) * Hc;
    float e = 0.f;
    for (int i = 0; i < Hc / 8; ++i) {
      h8 ev = *(const h8*)(ep + i * 8);
#pragma unroll
      for (int u = 0; u < 8; ++u) e += (float)ev[u] * hrow[i * 8 + u];
    }
    energ[(size_t)s * Bc + b0 + r] = e;
  }
}

// ---------------- decoder step part 2: softmax over BATCH axis (per s) ----------------
__global__ void __launch_bounds__(256) softmax_b_kernel(float* __restrict__ energ) {
  int s = blockIdx.x * blockDim.x + threadIdx.x;
  if (s >= Sc) return;
  float* row = energ + (size_t)s * Bc;  // contiguous over b
  float m = row[0];
  for (int b = 1; b < Bc; ++b) m = fmaxf(m, row[b]);
  float sum = 0.f;
  for (int b = 0; b < Bc; ++b) { float e = __expf(row[b] - m); row[b] = e; sum += e; }
  float inv = 1.f / sum;
  for (int b = 0; b < Bc; ++b) row[b] *= inv;
}

// ---------------- decoder step part 3: context + output linears + next pred ----------------
__global__ void __launch_bounds__(256)
dec_out_kernel(const float* __restrict__ energ, const _Float16* __restrict__ enc_out,
               const float* __restrict__ h2dec,
               const float* __restrict__ ww, const float* __restrict__ bw,
               const float* __restrict__ wop, const float* __restrict__ bop,
               float* __restrict__ pred, float* __restrict__ out_pred,
               float* __restrict__ out_ctx, int step) {
  __shared__ float wgt[16 * Sc];   // 32 KB attention weights for this row tile
  __shared__ float ctx[16 * Hc];   // 16 KB
  __shared__ float th[16 * DIc];
  const int tid = threadIdx.x;
  const int b0 = blockIdx.x * 16;

  for (int i = tid; i < 16 * Sc; i += 256) {
    int r = i >> 9, s = i & 511;
    wgt[i] = energ[(size_t)s * Bc + b0 + r];
  }
  __syncthreads();
  {  // context[b][h] = sum_s w[b][s] * enc[b][s][h]
    int j = tid;
    for (int r = 0; r < 16; ++r) {
      const _Float16* ep = enc_out + (size_t)(b0 + r) * Sc * Hc + j;
      const float* wr = &wgt[r * Sc];
      float a = 0.f;
      for (int s = 0; s < Sc; ++s) a += wr[s] * (float)ep[(size_t)s * Hc];
      ctx[r * Hc + j] = a;
      out_ctx[((size_t)(b0 + r) * Tc + step) * Hc + j] = a;
    }
  }
  __syncthreads();
  if (tid < 16 * DIc) {  // cats = [h2, ctx] @ ww^T + bw ; tanh
    int r = tid / DIc, oi = tid % DIc;
    const float* h2 = &h2dec[(size_t)(b0 + r) * Hc];
    const float* cr = &ctx[r * Hc];
    float a = bw[oi];
    for (int k = 0; k < Hc; ++k) a += h2[k] * ww[oi * 2 * Hc + k];
    for (int k = 0; k < Hc; ++k) a += cr[k] * ww[oi * 2 * Hc + Hc + k];
    th[tid] = tanhf(a);
  }
  __syncthreads();
  if (tid < 16 * DIc) {  // pred = tanh(cats) @ wop^T + bop
    int r = tid / DIc, oi = tid % DIc;
    float p = bop[oi];
#pragma unroll
    for (int k = 0; k < DIc; ++k) p += th[r * DIc + k] * wop[oi * DIc + k];
    out_pred[((size_t)(b0 + r) * Tc + step) * DIc + oi] = p;
    pred[(size_t)(b0 + r) * DIc + oi] = p;  // next step input
  }
}

// ---------------- host launcher ----------------
extern "C" void kernel_launch(void* const* d_in, const int* in_sizes, int n_in,
                              void* d_out, int out_size, void* d_ws, size_t ws_size,
                              hipStream_t stream) {
  (void)in_sizes; (void)n_in; (void)out_size; (void)ws_size;
  const float* srcs   = (const float*)d_in[0];
  const float* wih_e1 = (const float*)d_in[1];
  const float* whh_e1 = (const float*)d_in[2];
  const float* b_e1   = (const float*)d_in[3];
  const float* wih_e2 = (const float*)d_in[4];
  const float* whh_e2 = (const float*)d_in[5];
  const float* b_e2   = (const float*)d_in[6];
  const float* wdi    = (const float*)d_in[7];
  const float* bdi    = (const float*)d_in[8];
  const float* wih_d1 = (const float*)d_in[9];
  const float* whh_d1 = (const float*)d_in[10];
  const float* b_d1   = (const float*)d_in[11];
  const float* wih_d2 = (const float*)d_in[12];
  const float* whh_d2 = (const float*)d_in[13];
  const float* b_d2   = (const float*)d_in[14];
  const float* ww     = (const float*)d_in[15];
  const float* bw     = (const float*)d_in[16];
  const float* wop    = (const float*)d_in[17];
  const float* bop    = (const float*)d_in[18];

  char* ws = (char*)d_ws;
  size_t off = 0;
  auto take = [&](size_t bytes) -> void* {
    void* p = ws + off;
    off += (bytes + 255) & ~(size_t)255;
    return p;
  };
  const size_t WSZ = (size_t)Gc * Hc * sizeof(_Float16);  // 512 KB per swizzled matrix
  _Float16* s_whh_e1 = (_Float16*)take(WSZ);
  _Float16* s_wih_e2 = (_Float16*)take(WSZ);
  _Float16* s_whh_e2 = (_Float16*)take(WSZ);
  _Float16* s_wih_d1 = (_Float16*)take(WSZ);
  _Float16* s_whh_d1 = (_Float16*)take(WSZ);
  _Float16* s_wih_d2 = (_Float16*)take(WSZ);
  _Float16* encb  = (_Float16*)take((size_t)Bc * Sc * Hc * sizeof(_Float16));  // 64 MB, L2-resident
  float* hst   = (float*)take((size_t)Bc * Hc * sizeof(float));
  float* cst   = (float*)take((size_t)Bc * Hc * sizeof(float));
  float* h2d   = (float*)take((size_t)Bc * Hc * sizeof(float));
  float* energ = (float*)take((size_t)Sc * Bc * sizeof(float));
  float* pred  = (float*)take((size_t)Bc * DIc * sizeof(float));

  float* out_pred = (float*)d_out;                       // [B, T, 5]
  float* out_ctx  = out_pred + (size_t)Bc * Tc * DIc;    // [B, T, H]

  // weight prep (f16 swizzle into WMMA B-operand layout)
  swz_kernel<<<64, 256, 0, stream>>>(whh_e1, s_whh_e1);
  swz_kernel<<<64, 256, 0, stream>>>(wih_e2, s_wih_e2);
  swz_kernel<<<64, 256, 0, stream>>>(whh_e2, s_whh_e2);
  swz_kernel<<<64, 256, 0, stream>>>(wih_d1, s_wih_d1);
  swz_kernel<<<64, 256, 0, stream>>>(whh_d1, s_whh_d1);
  swz_kernel<<<64, 256, 0, stream>>>(wih_d2, s_wih_d2);
  init_pred_kernel<<<(Bc * DIc + 255) / 256, 256, 0, stream>>>(pred);

  // encoder: 16 batch tiles, recurrence lives in registers/LDS per WG
  enc_kernel<<<Bc / 16, 256, 0, stream>>>(srcs, wih_e1, b_e1, b_e2,
                                          s_whh_e1, s_wih_e2, s_whh_e2,
                                          encb, hst, cst);

  // decoder: 32 sequential steps; batch-axis softmax forces a kernel boundary
  for (int step = 0; step < Tc; ++step) {
    dec_cell_kernel<<<Bc / 16, 256, 0, stream>>>(wdi, bdi, b_d1, b_d2,
                                                 s_wih_d1, s_whh_d1, s_wih_d2,
                                                 encb, hst, cst, h2d, energ, pred);
    softmax_b_kernel<<<(Sc + 255) / 256, 256, 0, stream>>>(energ);
    dec_out_kernel<<<Bc / 16, 256, 0, stream>>>(energ, encb, h2d, ww, bw, wop, bop,
                                                pred, out_pred, out_ctx, step);
  }
}
